// HybridQLSTM_65481071398294
// MI455X (gfx1250) — matrix operand order repaired and checked by reading, not compiled, not run on hardware
//
#include <hip/hip_runtime.h>
#include <hip/hip_bf16.h>
#include <cstdint>

#define TT 2048
#define BSZ 64
#define DD 512
#define HH 512
#define DH 1024                       // D + H
#define KB_COUNT 32                   // DH / 32
#define CT_COUNT 32                   // H / 16
#define GSTRIDE (KB_COUNT * CT_COUNT * 512)   // bf16 elems per packed gate = 524288

typedef __attribute__((ext_vector_type(16))) __bf16 v16bf;
typedef __attribute__((ext_vector_type(8)))  float  v8f;

union ABf {
    v16bf v;
    uint4 q[2];
};

__device__ __forceinline__ float sigmoidf_(float x) {
    return 1.0f / (1.0f + __expf(-x));
}

// ---- Init 1: fp32 -> bf16 weight repack into WMMA B-fragment tile layout ----
// Packed index: idx = g*2^19 + kb*2^14 + ct*2^9 + lane*2^4 + i
// Element: k = kb*32 + (i>>3)*16 + (lane>>4)*8 + (i&7) ; n = ct*16 + (lane&15)
__global__ __launch_bounds__(256) void pack_weights(
    const float* __restrict__ Wf, const float* __restrict__ Wi,
    const float* __restrict__ Wg, const float* __restrict__ Wo,
    __bf16* __restrict__ Wp)
{
    unsigned int idx  = blockIdx.x * 256u + threadIdx.x;   // < 4 * GSTRIDE
    unsigned int i    = idx & 15u;
    unsigned int lane = (idx >> 4) & 31u;
    unsigned int ct   = (idx >> 9) & 31u;
    unsigned int kb   = (idx >> 14) & 31u;
    unsigned int g    = idx >> 19;
    unsigned int k = kb * 32u + (i >> 3) * 16u + (lane >> 4) * 8u + (i & 7u);     // 0..1023
    unsigned int n = ct * 16u + (lane & 15u);              // 0..511
    const float* W = (g == 0u) ? Wf : (g == 1u) ? Wi : (g == 2u) ? Wg : Wo;
    Wp[idx] = (__bf16)W[(size_t)k * HH + n];               // native v_cvt to bf16
}

// ---- Init 2: zero h0 (bf16) and c (fp32) ----
__global__ __launch_bounds__(256) void zero_state(__bf16* __restrict__ h0,
                                                  float* __restrict__ c)
{
    int idx = blockIdx.x * 256 + threadIdx.x;              // < B*H
    h0[idx] = (__bf16)0.0f;
    c[idx]  = 0.0f;
}

// Build a 16x32 bf16 A fragment from a row-major fp32 row pointer.
// ISA layout: elem e<8 -> K = cb + e ; e>=8 -> K = cb + 16 + (e-8), cb = (lane>>4)*8.
__device__ __forceinline__ v16bf a_frag_from_f32(const float* xr, int cb) {
    ABf a;
    float4 f0 = *(const float4*)(xr + cb);
    float4 f1 = *(const float4*)(xr + cb + 4);
    float4 f2 = *(const float4*)(xr + cb + 16);
    float4 f3 = *(const float4*)(xr + cb + 20);
    a.v[0]  = (__bf16)f0.x; a.v[1]  = (__bf16)f0.y; a.v[2]  = (__bf16)f0.z; a.v[3]  = (__bf16)f0.w;
    a.v[4]  = (__bf16)f1.x; a.v[5]  = (__bf16)f1.y; a.v[6]  = (__bf16)f1.z; a.v[7]  = (__bf16)f1.w;
    a.v[8]  = (__bf16)f2.x; a.v[9]  = (__bf16)f2.y; a.v[10] = (__bf16)f2.z; a.v[11] = (__bf16)f2.w;
    a.v[12] = (__bf16)f3.x; a.v[13] = (__bf16)f3.y; a.v[14] = (__bf16)f3.z; a.v[15] = (__bf16)f3.w;
    return a.v;
}

// ---- Per-step fused GEMM + LSTM cell ----
// Grid: 128 blocks (4 rowTiles x 32 colTiles), 256 threads = 8 waves.
// Wave w: gate = w>>1, K-half = w&1 (kh=0 -> x part, kh=1 -> h part).
__global__ __launch_bounds__(256) void lstm_step(
    const float* __restrict__ X,              // [T,B,D] fp32
    const __bf16* __restrict__ Wp,            // packed bf16 weights
    const float* __restrict__ bfv, const float* __restrict__ biv,
    const float* __restrict__ bgv, const float* __restrict__ bov,
    const __bf16* __restrict__ hin,           // [B,H] bf16
    __bf16* __restrict__ hout,                // [B,H] bf16
    float* __restrict__ c,                    // [B,H] fp32 (in-place)
    float* __restrict__ out,                  // [T,B,H] fp32
    int t)
{
    __shared__ float red[8][32][8];

    const int lane    = threadIdx.x & 31;
    const int wave    = threadIdx.x >> 5;
    const int gate    = wave >> 1;
    const int kh      = wave & 1;
    const int rowTile = blockIdx.x >> 5;     // 0..3
    const int colTile = blockIdx.x & 31;     // 0..31
    const int row     = rowTile * 16 + (lane & 15);  // A row this lane supplies
    const int half    = lane >> 4;           // 0 or 1 (K-interleave half)

    // Two accumulators (even/odd kb) -> consecutive WMMAs are independent,
    // hiding the WMMA->WMMA RAW hazard window.
    v8f accE = {};
    v8f accO = {};
    const __bf16* wg = Wp + (size_t)gate * GSTRIDE + (size_t)lane * 16;

    if (kh == 0) {
        // x contribution: kb = 0..15, fp32 -> bf16 via native converts.
        const float* xr = X + (size_t)t * (BSZ * DD) + (size_t)row * DD;
        #pragma unroll 2
        for (int kb = 0; kb < 16; kb += 2) {
            v16bf a0 = a_frag_from_f32(xr, kb * 32 + half * 8);
            v16bf a1 = a_frag_from_f32(xr, (kb + 1) * 32 + half * 8);
            const __bf16* bp0 = wg + ((size_t)kb * CT_COUNT + colTile) * 512;
            const __bf16* bp1 = wg + ((size_t)(kb + 1) * CT_COUNT + colTile) * 512;
            if (kb + 2 < 16)
                __builtin_prefetch(wg + ((size_t)(kb + 2) * CT_COUNT + colTile) * 512, 0, 1);
            v16bf b0 = *(const v16bf*)bp0;
            v16bf b1 = *(const v16bf*)bp1;
            accE = __builtin_amdgcn_wmma_f32_16x16x32_bf16(
                       false, a0, false, b0, (short)0, accE, false, false);
            accO = __builtin_amdgcn_wmma_f32_16x16x32_bf16(
                       false, a1, false, b1, (short)0, accO, false, false);
        }
    } else {
        // h contribution: kb = 16..31, h already bf16 in the ping-pong buffer.
        const __bf16* hr = hin + (size_t)row * HH;
        #pragma unroll 2
        for (int kb = 16; kb < 32; kb += 2) {
            ABf a0, a1;
            const int hb0 = (kb - 16) * 32 + half * 8;
            const int hb1 = hb0 + 32;
            a0.q[0] = *(const uint4*)(hr + hb0);
            a0.q[1] = *(const uint4*)(hr + hb0 + 16);
            a1.q[0] = *(const uint4*)(hr + hb1);
            a1.q[1] = *(const uint4*)(hr + hb1 + 16);
            const __bf16* bp0 = wg + ((size_t)kb * CT_COUNT + colTile) * 512;
            const __bf16* bp1 = wg + ((size_t)(kb + 1) * CT_COUNT + colTile) * 512;
            if (kb + 2 < 32)
                __builtin_prefetch(wg + ((size_t)(kb + 2) * CT_COUNT + colTile) * 512, 0, 1);
            v16bf b0 = *(const v16bf*)bp0;
            v16bf b1 = *(const v16bf*)bp1;
            accE = __builtin_amdgcn_wmma_f32_16x16x32_bf16(
                       false, a0.v, false, b0, (short)0, accE, false, false);
            accO = __builtin_amdgcn_wmma_f32_16x16x32_bf16(
                       false, a1.v, false, b1, (short)0, accO, false, false);
        }
    }

    #pragma unroll
    for (int j = 0; j < 8; ++j) red[wave][lane][j] = accE[j] + accO[j];
    __syncthreads();

    // Parallel epilogue: 256 threads <-> 256 tile elements. Thread (wave, lane)
    // handles element (m = (lane>>4)*8 + wave, n = lane&15): j index = wave.
    {
        const int j    = wave;
        const int n    = lane & 15;
        const int gcol = colTile * 16 + n;
        const int grow = rowTile * 16 + half * 8 + j;
        const float fpre = red[0][lane][j] + red[1][lane][j] + bfv[gcol];
        const float ipre = red[2][lane][j] + red[3][lane][j] + biv[gcol];
        const float gpre = red[4][lane][j] + red[5][lane][j] + bgv[gcol];
        const float opre = red[6][lane][j] + red[7][lane][j] + bov[gcol];
        const float fg = sigmoidf_(fpre);
        const float ig = sigmoidf_(ipre);
        const float gg = tanhf(gpre);
        const float og = sigmoidf_(opre);
        const size_t ci = (size_t)grow * HH + gcol;
        const float cn = fg * c[ci] + ig * gg;
        c[ci] = cn;
        const float hn = og * tanhf(cn);
        out[(size_t)t * (BSZ * HH) + ci] = hn;
        hout[ci] = (__bf16)hn;
    }
}

extern "C" void kernel_launch(void* const* d_in, const int* in_sizes, int n_in,
                              void* d_out, int out_size, void* d_ws, size_t ws_size,
                              hipStream_t stream)
{
    (void)in_sizes; (void)n_in; (void)out_size; (void)ws_size;

    const float* X   = (const float*)d_in[0];
    const float* Wf  = (const float*)d_in[1];
    const float* bf_ = (const float*)d_in[2];
    const float* Wi  = (const float*)d_in[3];
    const float* bi_ = (const float*)d_in[4];
    const float* Wg  = (const float*)d_in[5];
    const float* bg_ = (const float*)d_in[6];
    const float* Wo  = (const float*)d_in[7];
    const float* bo_ = (const float*)d_in[8];
    float* out = (float*)d_out;

    // Workspace layout (~4.3 MB):
    //   [0, 4MB)      packed bf16 weights (4 gates)
    //   [+0, 64KB)    h ping buffer (bf16)
    //   [+64KB,128KB) h pong buffer (bf16)
    //   [..,+128KB)   c state (fp32)
    uint8_t* ws = (uint8_t*)d_ws;
    __bf16* Wp = (__bf16*)ws;
    __bf16* h0 = (__bf16*)(ws + (size_t)4 * GSTRIDE * 2);
    __bf16* h1 = h0 + BSZ * HH;
    float* c = (float*)(h1 + BSZ * HH);

    pack_weights<<<dim3((4 * GSTRIDE) / 256), dim3(256), 0, stream>>>(Wf, Wi, Wg, Wo, Wp);
    zero_state<<<dim3((BSZ * HH) / 256), dim3(256), 0, stream>>>(h0, c);

    for (int t = 0; t < TT; ++t) {
        __bf16* hin  = (t & 1) ? h1 : h0;
        __bf16* hout = (t & 1) ? h0 : h1;
        lstm_step<<<dim3(128), dim3(256), 0, stream>>>(
            X, Wp, bf_, bi_, bg_, bo_, hin, hout, c, out, t);
    }

    // Tail of d_out: hT then cT.
    const size_t BH = (size_t)BSZ * HH;
    hipMemcpyAsync(out + (size_t)TT * BH, out + (size_t)(TT - 1) * BH,
                   BH * sizeof(float), hipMemcpyDeviceToDevice, stream);
    hipMemcpyAsync(out + (size_t)TT * BH + BH, c,
                   BH * sizeof(float), hipMemcpyDeviceToDevice, stream);
}
